// _MHA_68865505624599
// MI455X (gfx1250) — compile-verified, matrix operand order checked by reading, not run on hardware
//
#include <hip/hip_runtime.h>
#include <hip/hip_bf16.h>

#define DIM  1024
#define HEADS 16
#define DH   64
#define BATCH 4
#define SEQ  2048
#define BN   (BATCH * SEQ)   // 8192 token rows

typedef __attribute__((ext_vector_type(16))) __bf16 bf16x16;
typedef __attribute__((ext_vector_type(8)))  __bf16 bf16x8;
typedef __attribute__((ext_vector_type(8)))  float  f32x8;
typedef int v4i __attribute__((vector_size(16)));

#ifndef __has_builtin
#define __has_builtin(x) 0
#endif

#if __has_builtin(__builtin_amdgcn_global_load_async_to_lds_b128)
#define HAVE_ASYNC_LDS 1
#else
#define HAVE_ASYNC_LDS 0
#endif

__device__ __forceinline__ f32x8 wmma_bf16(bf16x16 a, bf16x16 b, f32x8 c) {
    // 8 args: (neg_a, A, neg_b, B, c_mod, C, reuse_a, reuse_b)
    return __builtin_amdgcn_wmma_f32_16x16x32_bf16(false, a, false, b, (short)0, c, false, false);
}

// Load a 16x32 bf16 A/B fragment per the CDNA5 ISA VGPR layout:
// lane l: row = l&15, half = l>>4; vgprs 0-3 hold K = 8*half..8*half+7,
// vgprs 4-7 hold K = 16+8*half..16+8*half+7  -> two contiguous 16B loads.
__device__ __forceinline__ bf16x16 load_frag(const __bf16* base, int row, int ld, int k0) {
    const int l = threadIdx.x & 31;
    const int r = l & 15;
    const int half = l >> 4;
    const __bf16* p = base + (size_t)(row + r) * (size_t)ld + k0 + 8 * half;
    bf16x8 lo = *(const bf16x8*)(p);
    bf16x8 hi = *(const bf16x8*)(p + 16);
    bf16x16 f;
#pragma unroll
    for (int i = 0; i < 8; ++i) { f[i] = lo[i]; f[i + 8] = hi[i]; }
    return f;
}

// 16-byte global -> LDS copy: async DMA path on gfx1250 if the builtin exists.
__device__ __forceinline__ void cp16(const __bf16* g, __bf16* l) {
#if HAVE_ASYNC_LDS
    __attribute__((address_space(1))) v4i* gp =
        (__attribute__((address_space(1))) v4i*)(uintptr_t)g;
    __attribute__((address_space(3))) __bf16* l3 =
        (__attribute__((address_space(3))) __bf16*)l;               // addrspacecast
    __attribute__((address_space(3))) v4i* lp =
        (__attribute__((address_space(3))) v4i*)l3;                 // pointee reinterpret
    __builtin_amdgcn_global_load_async_to_lds_b128(gp, lp, 0, 0);
#else
    *(bf16x8*)l = *(const bf16x8*)g;
#endif
}

__device__ __forceinline__ void wait_async_copies() {
#if HAVE_ASYNC_LDS
    asm volatile("s_wait_asynccnt 0x0" ::: "memory");
#endif
}

// ---------------- prep kernels ----------------

__global__ void k_cvt_bf16(const float* __restrict__ src, __bf16* __restrict__ dst, int n) {
    int i = blockIdx.x * 256 + threadIdx.x;
    if (i < n) dst[i] = (__bf16)src[i];
}

// src[rows][cols] fp32 -> dst[cols][rows] bf16
__global__ void k_transpose_cvt(const float* __restrict__ src, __bf16* __restrict__ dst,
                                int rows, int cols) {
    int i = blockIdx.x * 256 + threadIdx.x;
    if (i < rows * cols) {
        int k = i / cols, c = i - k * cols;
        dst[(size_t)c * rows + k] = (__bf16)src[i];
    }
}

// ---------------- QKV GEMM:  qkv = x @ W_qkv + b ; scatter into Q,K,Vt ----------------
// grid (3072/128, 8192/128), 256 threads = 8 waves; wave tile 64x32 (4x2 WMMA accs)
__global__ void __launch_bounds__(256)
k_qkv(const __bf16* __restrict__ xb, const __bf16* __restrict__ wT,
      const float* __restrict__ bias,
      __bf16* __restrict__ Q, __bf16* __restrict__ K, __bf16* __restrict__ Vt) {
    const int l  = threadIdx.x & 31;
    const int w  = threadIdx.x >> 5;
    const int wm = w >> 2, wn = w & 3;
    const int rowBase = blockIdx.y * 128 + wm * 64;
    const int colBase = blockIdx.x * 128 + wn * 32;

    f32x8 acc[4][2] = {};
    for (int kk = 0; kk < DIM; kk += 32) {
        bf16x16 a0 = load_frag(xb, rowBase,      DIM, kk);
        bf16x16 a1 = load_frag(xb, rowBase + 16, DIM, kk);
        bf16x16 a2 = load_frag(xb, rowBase + 32, DIM, kk);
        bf16x16 a3 = load_frag(xb, rowBase + 48, DIM, kk);
        bf16x16 b0 = load_frag(wT, colBase,      DIM, kk);
        bf16x16 b1 = load_frag(wT, colBase + 16, DIM, kk);
        acc[0][0] = wmma_bf16(a0, b0, acc[0][0]);
        acc[0][1] = wmma_bf16(a0, b1, acc[0][1]);
        acc[1][0] = wmma_bf16(a1, b0, acc[1][0]);
        acc[1][1] = wmma_bf16(a1, b1, acc[1][1]);
        acc[2][0] = wmma_bf16(a2, b0, acc[2][0]);
        acc[2][1] = wmma_bf16(a2, b1, acc[2][1]);
        acc[3][0] = wmma_bf16(a3, b0, acc[3][0]);
        acc[3][1] = wmma_bf16(a3, b1, acc[3][1]);
    }

    const int half = l >> 4;
#pragma unroll
    for (int j = 0; j < 2; ++j) {
        const int c = colBase + j * 16 + (l & 15);     // qkv column in [0,3072)
        const int which = c >> 10;                     // 0=q 1=k 2=v
        const int cc = c & 1023;
        const int h = cc >> 6, d = cc & 63;
        const float bv = bias[c];
#pragma unroll
        for (int i = 0; i < 4; ++i) {
#pragma unroll
            for (int r = 0; r < 8; ++r) {
                const int m = rowBase + i * 16 + r + 8 * half;  // token row [0,8192)
                const int b = m >> 11, n = m & 2047;
                const float v = acc[i][j][r] + bv;
                if (which == 0)
                    Q[((size_t)(b * HEADS + h) * SEQ + n) * DH + d] = (__bf16)v;
                else if (which == 1)
                    K[((size_t)(b * HEADS + h) * SEQ + n) * DH + d] = (__bf16)v;
                else
                    Vt[((size_t)(b * HEADS + h) * DH + d) * SEQ + n] = (__bf16)v;
            }
        }
    }
}

// ---------------- Flash attention ----------------
// grid (SEQ/128, HEADS, BATCH), 256 threads = 8 waves; wave owns 16 query rows.
// K/V tiles staged cooperatively into LDS (async DMA, double buffered).
__global__ void __launch_bounds__(256)
k_attn(const __bf16* __restrict__ Q, const __bf16* __restrict__ K,
       const __bf16* __restrict__ Vt, __bf16* __restrict__ ctx) {
    __shared__ __bf16 sK[2][32 * DH];   // 32 keys x 64 dims, 4KB per buffer
    __shared__ __bf16 sV[2][DH * 32];   // 64 dims x 32 keys (V^T rows), 4KB per buffer
    __shared__ __bf16 sP[8 * 512];      // per-wave 16x32 P staging (1KB/wave)

    const int tid  = threadIdx.x;
    const int l    = tid & 31;
    const int w    = tid >> 5;
    const int half = l >> 4;
    const int b = blockIdx.z, h = blockIdx.y;
    const int qBase = blockIdx.x * 128 + w * 16;

    const __bf16* Qh = Q  + (size_t)(b * HEADS + h) * SEQ * DH;
    const __bf16* Kh = K  + (size_t)(b * HEADS + h) * SEQ * DH;
    const __bf16* Vh = Vt + (size_t)(b * HEADS + h) * DH * SEQ;
    __bf16* sPw = sP + w * 512;

    // per-thread staging coordinates (256 threads x 16B == one 4KB tile each)
    const int krow = tid >> 3, kseg = (tid & 7) * 8;   // K: 32 rows x 64 cols
    const int vrow = tid >> 2, vseg = (tid & 3) * 8;   // V: 64 rows x 32 cols

    // stage tile 0
    cp16(Kh + (size_t)krow * DH + kseg, &sK[0][krow * DH + kseg]);
    cp16(Vh + (size_t)vrow * SEQ + vseg, &sV[0][vrow * 32 + vseg]);

    bf16x16 q0 = load_frag(Qh, qBase, DH, 0);
    bf16x16 q1 = load_frag(Qh, qBase, DH, 32);

    float mvec[8], lvec[8];
#pragma unroll
    for (int r = 0; r < 8; ++r) { mvec[r] = -3.0e38f; lvec[r] = 0.0f; }
    f32x8 o[4] = {};

    const float scale = 0.125f;   // 1/sqrt(64)

    for (int kb = 0, it = 0; kb < SEQ; kb += 32, ++it) {
        const int buf = it & 1;
        wait_async_copies();
        __syncthreads();
        if (kb + 32 < SEQ) {   // prefetch next K/V tile into the other buffer
            cp16(Kh + (size_t)(kb + 32 + krow) * DH + kseg, &sK[buf ^ 1][krow * DH + kseg]);
            cp16(Vh + (size_t)vrow * SEQ + (kb + 32) + vseg, &sV[buf ^ 1][vrow * 32 + vseg]);
        }

        // S = Q @ K^T for 32 keys (two 16x16 score tiles)
        bf16x16 k0a = load_frag(sK[buf], 0,  DH, 0);
        bf16x16 k0b = load_frag(sK[buf], 0,  DH, 32);
        bf16x16 k1a = load_frag(sK[buf], 16, DH, 0);
        bf16x16 k1b = load_frag(sK[buf], 16, DH, 32);
        f32x8 s0 = {}, s1 = {};
        s0 = wmma_bf16(q0, k0a, s0);
        s0 = wmma_bf16(q1, k0b, s0);
        s1 = wmma_bf16(q0, k1a, s1);
        s1 = wmma_bf16(q1, k1b, s1);

        // online softmax; rows live at (vgpr r, half), cols across 16-lane groups
#pragma unroll
        for (int r = 0; r < 8; ++r) {
            float x0 = s0[r] * scale, x1 = s1[r] * scale;
            float t = fmaxf(x0, x1);
            t = fmaxf(t, __shfl_xor(t, 1, 32));
            t = fmaxf(t, __shfl_xor(t, 2, 32));
            t = fmaxf(t, __shfl_xor(t, 4, 32));
            t = fmaxf(t, __shfl_xor(t, 8, 32));
            const float mnew  = fmaxf(mvec[r], t);
            const float alpha = __expf(mvec[r] - mnew);
            const float p0 = __expf(x0 - mnew);
            const float p1 = __expf(x1 - mnew);
            float rs = p0 + p1;
            rs += __shfl_xor(rs, 1, 32);
            rs += __shfl_xor(rs, 2, 32);
            rs += __shfl_xor(rs, 4, 32);
            rs += __shfl_xor(rs, 8, 32);
            lvec[r] = lvec[r] * alpha + rs;
            mvec[r] = mnew;
#pragma unroll
            for (int t4 = 0; t4 < 4; ++t4) o[t4][r] *= alpha;
            // stage P (C/D layout -> row-major 16x32 in LDS)
            const int m = r + 8 * half;
            sPw[m * 32 + (l & 15)]      = (__bf16)p0;
            sPw[m * 32 + 16 + (l & 15)] = (__bf16)p1;
        }
        asm volatile("s_wait_dscnt 0x0" ::: "memory");

        // O += P @ V
        bf16x16 pf = load_frag(sPw, 0, 32, 0);
        bf16x16 v0 = load_frag(sV[buf], 0,  32, 0);
        bf16x16 v1 = load_frag(sV[buf], 16, 32, 0);
        bf16x16 v2 = load_frag(sV[buf], 32, 32, 0);
        bf16x16 v3 = load_frag(sV[buf], 48, 32, 0);
        o[0] = wmma_bf16(pf, v0, o[0]);
        o[1] = wmma_bf16(pf, v1, o[1]);
        o[2] = wmma_bf16(pf, v2, o[2]);
        o[3] = wmma_bf16(pf, v3, o[3]);
    }

    // normalize + write context rows [b*SEQ + tok][h*64 + d]
#pragma unroll
    for (int r = 0; r < 8; ++r) {
        const float inv = 1.0f / lvec[r];
        const int tok = qBase + r + 8 * half;
        const size_t rowOff = ((size_t)b * SEQ + tok) * DIM + h * DH;
#pragma unroll
        for (int t4 = 0; t4 < 4; ++t4)
            ctx[rowOff + t4 * 16 + (l & 15)] = (__bf16)(o[t4][r] * inv);
    }
}

// ---------------- Projection GEMM: out = ctx @ W_proj + b (fp32 out) ----------------
// grid (1024/128, 8192/128), wave tile 64x32
__global__ void __launch_bounds__(256)
k_proj(const __bf16* __restrict__ a, const __bf16* __restrict__ wT,
       const float* __restrict__ bias, float* __restrict__ out) {
    const int l  = threadIdx.x & 31;
    const int w  = threadIdx.x >> 5;
    const int wm = w >> 2, wn = w & 3;
    const int rowBase = blockIdx.y * 128 + wm * 64;
    const int colBase = blockIdx.x * 128 + wn * 32;

    f32x8 acc[4][2] = {};
    for (int kk = 0; kk < DIM; kk += 32) {
        bf16x16 a0 = load_frag(a, rowBase,      DIM, kk);
        bf16x16 a1 = load_frag(a, rowBase + 16, DIM, kk);
        bf16x16 a2 = load_frag(a, rowBase + 32, DIM, kk);
        bf16x16 a3 = load_frag(a, rowBase + 48, DIM, kk);
        bf16x16 b0 = load_frag(wT, colBase,      DIM, kk);
        bf16x16 b1 = load_frag(wT, colBase + 16, DIM, kk);
        acc[0][0] = wmma_bf16(a0, b0, acc[0][0]);
        acc[0][1] = wmma_bf16(a0, b1, acc[0][1]);
        acc[1][0] = wmma_bf16(a1, b0, acc[1][0]);
        acc[1][1] = wmma_bf16(a1, b1, acc[1][1]);
        acc[2][0] = wmma_bf16(a2, b0, acc[2][0]);
        acc[2][1] = wmma_bf16(a2, b1, acc[2][1]);
        acc[3][0] = wmma_bf16(a3, b0, acc[3][0]);
        acc[3][1] = wmma_bf16(a3, b1, acc[3][1]);
    }

    const int half = l >> 4;
#pragma unroll
    for (int j = 0; j < 2; ++j) {
        const int c = colBase + j * 16 + (l & 15);
        const float bv = bias[c];
#pragma unroll
        for (int i = 0; i < 4; ++i) {
#pragma unroll
            for (int r = 0; r < 8; ++r) {
                const int m = rowBase + i * 16 + r + 8 * half;
                out[(size_t)m * DIM + c] = acc[i][j][r] + bv;
            }
        }
    }
}

// ---------------- launcher ----------------
extern "C" void kernel_launch(void* const* d_in, const int* in_sizes, int n_in,
                              void* d_out, int out_size, void* d_ws, size_t ws_size,
                              hipStream_t stream) {
    (void)in_sizes; (void)n_in; (void)out_size; (void)ws_size;
    const float* x  = (const float*)d_in[0];
    const float* Wq = (const float*)d_in[1];
    const float* bq = (const float*)d_in[2];
    const float* Wp = (const float*)d_in[3];
    const float* bp = (const float*)d_in[4];
    float* out = (float*)d_out;

    // workspace layout (bf16 elements): xb | WqkvT | WprojT | Q | K | Vt | ctx  (~88 MB)
    __bf16* ws = (__bf16*)d_ws;
    const size_t SZ_X   = (size_t)BN * DIM;                 // 8Mi
    const size_t SZ_WQ  = (size_t)3 * DIM * DIM;            // 3Mi
    const size_t SZ_WP  = (size_t)DIM * DIM;                // 1Mi
    const size_t SZ_QKV = (size_t)BATCH * HEADS * SEQ * DH; // 8Mi
    __bf16* xb  = ws;
    __bf16* wqT = xb  + SZ_X;
    __bf16* wpT = wqT + SZ_WQ;
    __bf16* Qb  = wpT + SZ_WP;
    __bf16* Kb  = Qb  + SZ_QKV;
    __bf16* Vt  = Kb  + SZ_QKV;
    __bf16* ctx = Vt  + SZ_QKV;

    k_cvt_bf16<<<(unsigned)((SZ_X + 255) / 256), 256, 0, stream>>>(x, xb, (int)SZ_X);
    k_transpose_cvt<<<(unsigned)((SZ_WQ + 255) / 256), 256, 0, stream>>>(Wq, wqT, DIM, 3 * DIM);
    k_transpose_cvt<<<(unsigned)((SZ_WP + 255) / 256), 256, 0, stream>>>(Wp, wpT, DIM, DIM);

    k_qkv <<<dim3(3 * DIM / 128, BN / 128), 256, 0, stream>>>(xb, wqT, bq, Qb, Kb, Vt);
    k_attn<<<dim3(SEQ / 128, HEADS, BATCH), 256, 0, stream>>>(Qb, Kb, Vt, ctx);
    k_proj<<<dim3(DIM / 128, BN / 128), 256, 0, stream>>>(ctx, wpT, bp, out);
}
